// LinformerHead_11458972746304
// MI455X (gfx1250) — matrix operand order repaired and checked by reading, not compiled, not run on hardware
//
#include <hip/hip_runtime.h>

#define HEAD  64
#define TSEQ  2048
#define CEMB  1024
#define NBAT  4

typedef __attribute__((ext_vector_type(16))) __bf16 v16bf;
typedef __attribute__((ext_vector_type(8)))  __bf16 v8bf;
typedef __attribute__((ext_vector_type(8)))  float  v8f;
typedef __attribute__((ext_vector_type(4)))  float  v4f;

static __device__ __forceinline__ v16bf cat8(v8bf lo, v8bf hi) {
    return __builtin_shufflevector(lo, hi, 0,1,2,3,4,5,6,7,8,9,10,11,12,13,14,15);
}

static __device__ __forceinline__ v8f wmma_bf16(v16bf a, v16bf b, v8f c) {
    return __builtin_amdgcn_wmma_f32_16x16x32_bf16(
        /*neg_a=*/false, a, /*neg_b=*/false, b,
        /*c_mod=*/(short)0, c, /*reuse_a=*/false, /*reuse_b=*/false);
}

// ---------------------------------------------------------------------------
// Kernel 0: one-shot fp32 -> bf16 conversion (x and the three W matrices).
// 8 elements per thread, vectorized 32B loads / 16B stores.
// ---------------------------------------------------------------------------
__global__ __launch_bounds__(256) void cvt_bf16_kernel(
    const float* __restrict__ in, __bf16* __restrict__ out, int n8)
{
    const int i = blockIdx.x * blockDim.x + threadIdx.x;
    if (i >= n8) return;
    const v4f* p = (const v4f*)(in + (size_t)i * 8);
    v4f a = p[0], b = p[1];
    v8bf o;
    #pragma unroll
    for (int j = 0; j < 4; ++j) { o[j] = (__bf16)a[j]; o[4 + j] = (__bf16)b[j]; }
    *(v8bf*)(out + (size_t)i * 8) = o;
}

// ---------------------------------------------------------------------------
// Kernel 1: fused QKV projection from pre-converted bf16 operands.
// One wave per (matrix, 16-row tile of T).  Inner loop: pure loads + WMMA.
// q,k stored bf16 [B,T,64] (q pre-scaled by 1/sqrt(H)); v stored bf16 [B,64,T].
// ---------------------------------------------------------------------------
__global__ __launch_bounds__(128) void qkv_proj_kernel(
    const __bf16* __restrict__ xbf,
    const __bf16* __restrict__ Wqb, const __bf16* __restrict__ Wkb,
    const __bf16* __restrict__ Wvb,
    __bf16* __restrict__ qbuf, __bf16* __restrict__ kbuf,
    __bf16* __restrict__ vT)
{
    const int wid  = blockIdx.x * (blockDim.x >> 5) + (threadIdx.x >> 5);
    const int lane = threadIdx.x & 31;
    const int half = lane >> 4;      // 0: lanes 0-15, 1: lanes 16-31
    const int r    = lane & 15;

    const int mat  = wid % 3;                 // 0=q, 1=k, 2=v
    const int tile = wid / 3;
    if (tile >= NBAT * (TSEQ / 16)) return;
    const int b  = tile / (TSEQ / 16);
    const int t0 = (tile % (TSEQ / 16)) * 16;

    const __bf16* W = (mat == 0) ? Wqb : (mat == 1) ? Wkb : Wvb;
    const __bf16* xrow = xbf + ((size_t)b * TSEQ + t0 + r) * CEMB;

    v8f acc[4] = {};

    for (int c0 = 0; c0 < CEMB; c0 += 32) {
        // A fragment (16-bit 16x32 layout): lane holds row t0+r,
        // K = {abase..abase+7, abase+16..abase+23}, abase = c0 + half*8
        const int abase = c0 + half * 8;
        v16bf afrag = cat8(*(const v8bf*)(xrow + abase),
                           *(const v8bf*)(xrow + abase + 16));

        // B fragments: B = W^T (K=c dim, N=h dim); lane holds col h = nt*16+r,
        // K = {kbase..kbase+7, kbase+16..kbase+23}, kbase = c0 + half*8
        const int kbase = c0 + half * 8;
        #pragma unroll
        for (int nt = 0; nt < 4; ++nt) {
            const __bf16* wrow = W + (size_t)(nt * 16 + r) * CEMB + kbase;
            v16bf bfrag = cat8(*(const v8bf*)wrow, *(const v8bf*)(wrow + 16));
            acc[nt] = wmma_bf16(afrag, bfrag, acc[nt]);
        }
    }

    // ---- store ----
    const float scale = (mat == 0) ? 0.125f : 1.0f;   // 1/sqrt(64) folded into q
    if (mat < 2) {
        __bf16* dst = (mat == 0) ? qbuf : kbuf;
        #pragma unroll
        for (int nt = 0; nt < 4; ++nt)
            #pragma unroll
            for (int i = 0; i < 8; ++i) {
                const int trow = t0 + half * 8 + i;           // C/D: M = half*8+i
                dst[((size_t)b * TSEQ + trow) * HEAD + nt * 16 + r] =
                    (__bf16)(acc[nt][i] * scale);
            }
    } else {
        // vT[b][h][t]: per lane h = nt*16 + r fixed, 8 consecutive t -> 16B store
        #pragma unroll
        for (int nt = 0; nt < 4; ++nt) {
            v8bf pk;
            #pragma unroll
            for (int i = 0; i < 8; ++i) pk[i] = (__bf16)acc[nt][i];
            __bf16* dst = vT + ((size_t)b * HEAD + nt * 16 + r) * TSEQ + t0 + half * 8;
            *(v8bf*)dst = pk;
        }
    }
}

// ---------------------------------------------------------------------------
// Kernel 2: flash attention. One wave per 16-query tile, 32-key blocks.
// ---------------------------------------------------------------------------
__global__ __launch_bounds__(256) void flash_attn_kernel(
    const __bf16* __restrict__ qbuf, const __bf16* __restrict__ kbuf,
    const __bf16* __restrict__ vT, float* __restrict__ out)
{
    __shared__ __bf16 lds[8][16 * 32];      // per-wave 16x32 P tile (8 KB)

    const int wslot = threadIdx.x >> 5;
    const int wid   = blockIdx.x * 8 + wslot;
    const int lane  = threadIdx.x & 31;
    const int half  = lane >> 4;
    const int r     = lane & 15;

    const int b  = wid / (TSEQ / 16);
    const int q0 = (wid % (TSEQ / 16)) * 16;

    // q A-fragments for both 32-wide K(=H) chunks, resident across the loop
    const __bf16* qrow = qbuf + ((size_t)b * TSEQ + q0 + r) * HEAD;
    v16bf qa[2];
    #pragma unroll
    for (int c = 0; c < 2; ++c) {
        const int base = c * 32 + half * 8;
        qa[c] = cat8(*(const v8bf*)(qrow + base),
                     *(const v8bf*)(qrow + base + 16));
    }

    float m[8], l[8];
    #pragma unroll
    for (int i = 0; i < 8; ++i) { m[i] = -__builtin_inff(); l[i] = 0.0f; }
    v8f acc[4] = {};

    const int nblocks = (q0 + 15) / 32 + 1;
    for (int kb = 0; kb < nblocks; ++kb) {
        const int s0 = kb * 32;

        // ---- S = q @ k^T (16x32 tile, two N-subtiles of 16 keys) ----
        v8f S[2] = {};
        #pragma unroll
        for (int u = 0; u < 2; ++u) {
            #pragma unroll
            for (int c = 0; c < 2; ++c) {
                const __bf16* krow = kbuf +
                    ((size_t)b * TSEQ + s0 + u * 16 + r) * HEAD + c * 32 + half * 8;
                v16bf kf = cat8(*(const v8bf*)krow, *(const v8bf*)(krow + 16));
                S[u] = wmma_bf16(qa[c], kf, S[u]);
            }
        }

        // ---- causal mask (only diagonal-touching blocks) ----
        if (s0 + 31 > q0) {
            #pragma unroll
            for (int u = 0; u < 2; ++u)
                #pragma unroll
                for (int i = 0; i < 8; ++i) {
                    const int s = s0 + u * 16 + r;
                    const int q = q0 + half * 8 + i;
                    if (s > q) S[u][i] = -__builtin_inff();
                }
        }

        // ---- online softmax (fp32); rows 0-7 in lanes 0-15, 8-15 in 16-31 ----
        #pragma unroll
        for (int i = 0; i < 8; ++i) {
            float t = fmaxf(S[0][i], S[1][i]);
            t = fmaxf(t, __shfl_xor(t, 8, 32));
            t = fmaxf(t, __shfl_xor(t, 4, 32));
            t = fmaxf(t, __shfl_xor(t, 2, 32));
            t = fmaxf(t, __shfl_xor(t, 1, 32));
            const float mn    = fmaxf(m[i], t);
            const float alpha = __expf(m[i] - mn);
            const float p0    = __expf(S[0][i] - mn);
            const float p1    = __expf(S[1][i] - mn);
            float rs = p0 + p1;
            rs += __shfl_xor(rs, 8, 32);
            rs += __shfl_xor(rs, 4, 32);
            rs += __shfl_xor(rs, 2, 32);
            rs += __shfl_xor(rs, 1, 32);
            l[i] = l[i] * alpha + rs;
            m[i] = mn;
            #pragma unroll
            for (int nt = 0; nt < 4; ++nt) acc[nt][i] *= alpha;
            // stash P (bf16) in C/D layout: row = half*8+i, cols r and 16+r
            lds[wslot][(half * 8 + i) * 32 + r]      = (__bf16)p0;
            lds[wslot][(half * 8 + i) * 32 + 16 + r] = (__bf16)p1;
        }

        // ---- re-read P in A-fragment layout (row = lane&15) ----
        const __bf16* pl = &lds[wslot][r * 32];
        v16bf pa = cat8(*(const v8bf*)(pl + half * 8),
                        *(const v8bf*)(pl + 16 + half * 8));

        // ---- acc += P @ V  (B-frags contiguous thanks to vT layout) ----
        #pragma unroll
        for (int nt = 0; nt < 4; ++nt) {
            const __bf16* vrow = vT +
                ((size_t)b * HEAD + nt * 16 + r) * TSEQ + s0 + half * 8;
            v16bf vf = cat8(*(const v8bf*)vrow, *(const v8bf*)(vrow + 16));
            acc[nt] = wmma_bf16(pa, vf, acc[nt]);
        }
    }

    // ---- epilogue: normalize and store fp32 [B,T,64] ----
    #pragma unroll
    for (int i = 0; i < 8; ++i) {
        const float inv = 1.0f / l[i];
        const int trow = q0 + half * 8 + i;
        #pragma unroll
        for (int nt = 0; nt < 4; ++nt)
            out[((size_t)b * TSEQ + trow) * HEAD + nt * 16 + r] = acc[nt][i] * inv;
    }
}

// ---------------------------------------------------------------------------
extern "C" void kernel_launch(void* const* d_in, const int* in_sizes, int n_in,
                              void* d_out, int out_size, void* d_ws, size_t ws_size,
                              hipStream_t stream) {
    const float* x  = (const float*)d_in[0];
    const float* Wq = (const float*)d_in[1];
    const float* Wk = (const float*)d_in[2];
    const float* Wv = (const float*)d_in[3];
    float* out = (float*)d_out;

    const size_t NX = (size_t)NBAT * TSEQ * CEMB;   // x elements   (8.4M)
    const size_t NW = (size_t)HEAD * CEMB;          // W elements   (64K)
    const size_t NQ = (size_t)NBAT * TSEQ * HEAD;   // q/k/v elems  (512K)

    __bf16* xbf  = (__bf16*)d_ws;
    __bf16* Wqb  = xbf + NX;
    __bf16* Wkb  = Wqb + NW;
    __bf16* Wvb  = Wkb + NW;
    __bf16* qbuf = Wvb + NW;
    __bf16* kbuf = qbuf + NQ;
    __bf16* vT   = kbuf + NQ;

    // --- one-shot bf16 conversion (8 elems/thread) ---
    cvt_bf16_kernel<<<(int)(NX / 8 / 256), 256, 0, stream>>>(x, xbf, (int)(NX / 8));
    cvt_bf16_kernel<<<(int)(NW / 8 / 256), 256, 0, stream>>>(Wq, Wqb, (int)(NW / 8));
    cvt_bf16_kernel<<<(int)(NW / 8 / 256), 256, 0, stream>>>(Wk, Wkb, (int)(NW / 8));
    cvt_bf16_kernel<<<(int)(NW / 8 / 256), 256, 0, stream>>>(Wv, Wvb, (int)(NW / 8));

    // 512 tiles x 3 matrices = 1536 waves; 4 waves/block -> 384 blocks
    qkv_proj_kernel<<<384, 128, 0, stream>>>(xbf, Wqb, Wkb, Wvb, qbuf, kbuf, vT);

    // 512 query tiles, 8 waves/block -> 64 blocks
    flash_attn_kernel<<<64, 256, 0, stream>>>(qbuf, kbuf, vT, out);
}
